// MultiHeadAttention_87222195847929
// MI455X (gfx1250) — matrix operand order repaired and checked by reading, not compiled, not run on hardware
//
#include <hip/hip_runtime.h>
#include <hip/hip_bf16.h>

// ---------------------------------------------------------------------------
// MI455X / gfx1250 multi-head banded attention, bf16 WMMA path (wave32).
// ---------------------------------------------------------------------------

typedef __attribute__((ext_vector_type(16))) __bf16 v16bf;
typedef __attribute__((ext_vector_type(8)))  float  v8f;

union FragU {
    v16bf v;
    uint4 q[2];
    unsigned short s[16];
};

__device__ inline unsigned short f2bfu(float f) {
    union { float f; unsigned u; } x;
    x.f = f;
    unsigned u = x.u + 0x7FFFu + ((x.u >> 16) & 1u);   // round-to-nearest-even
    return (unsigned short)(u >> 16);
}

// A-fragment (16x32 bf16): lane holds row M = lane&15; K runs
// [k0+half*8, +8) and [k0+16+half*8, +8)  (ISA 7.12.2, 16-bit A layout)
__device__ inline v16bf load_a(const unsigned short* base, int row, int ld,
                               int k0, int half) {
    FragU f;
    const unsigned short* p = base + (size_t)row * ld + k0 + half * 8;
    f.q[0] = *(const uint4*)(p);
    f.q[1] = *(const uint4*)(p + 16);
    return f.v;
}

// B-fragment (32x16 bf16): lane holds column N = lane&15; K runs
// [k0+half*8, +8) and [k0+16+half*8, +8)  (same K order as A fragment)
__device__ inline v16bf load_b(const unsigned short* base, int col, int ld,
                               int k0, int half) {
    FragU f;
    const unsigned short* p = base + (size_t)col * ld + k0 + half * 8;
    f.q[0] = *(const uint4*)(p);
    f.q[1] = *(const uint4*)(p + 16);
    return f.v;
}

__device__ inline v8f wmma_bf16(v16bf a, v16bf b, v8f c) {
    return __builtin_amdgcn_wmma_f32_16x16x32_bf16(
        /*neg_a=*/false, a, /*neg_b=*/false, b,
        /*c_mod=*/(short)0, c, /*reuse_a=*/false, /*reuse_b=*/false);
}

// ---------------------------------------------------------------------------
// Conversions
// ---------------------------------------------------------------------------
__global__ void cvt_x_kernel(const float* __restrict__ X,
                             unsigned short* __restrict__ Xbf, int n) {
    int i = blockIdx.x * blockDim.x + threadIdx.x;
    if (i < n) Xbf[i] = f2bfu(X[i]);
}

// W[k][n] (1024x1024 f32) -> Wt[n][k] bf16
__global__ void cvt_wT_kernel(const float* __restrict__ W,
                              unsigned short* __restrict__ Wt) {
    int i = blockIdx.x * blockDim.x + threadIdx.x;
    int k = i >> 10;
    int n = i & 1023;
    Wt[(size_t)n * 1024 + k] = f2bfu(W[i]);
}

// ---------------------------------------------------------------------------
// Projection GEMM: out = Xbf[4096x1024] @ Wt^T + bias, stored per-head bf16.
// One wave computes a 32x64 output tile: 2 A-frags x 4 B-frags -> 8 WMMAs
// per K-step (each B fragment reused twice -> ~11 FLOP/byte from cache).
// transposeV=0 -> [B,H,S,Dh] ; transposeV=1 -> [B,H,Dh,S]
// ---------------------------------------------------------------------------
__global__ void proj_gemm_kernel(const unsigned short* __restrict__ Xbf,
                                 const unsigned short* __restrict__ Wt,
                                 const float* __restrict__ bias,
                                 unsigned short* __restrict__ out,
                                 int transposeV) {
    const int m0 = blockIdx.x * 32;
    const int n0 = blockIdx.y * 64;
    const int lane = threadIdx.x;
    const int half = lane >> 4;
    const int lr = lane & 15;

    v8f cz = {0.f,0.f,0.f,0.f,0.f,0.f,0.f,0.f};
    v8f c00 = cz, c01 = cz, c02 = cz, c03 = cz;   // rows m0..m0+15
    v8f c10 = cz, c11 = cz, c12 = cz, c13 = cz;   // rows m0+16..m0+31

    for (int k0 = 0; k0 < 1024; k0 += 32) {
        v16bf a0 = load_a(Xbf, m0 +      lr, 1024, k0, half);
        v16bf a1 = load_a(Xbf, m0 + 16 + lr, 1024, k0, half);
        v16bf b0 = load_b(Wt, n0 +  0 + lr, 1024, k0, half);
        v16bf b1 = load_b(Wt, n0 + 16 + lr, 1024, k0, half);
        v16bf b2 = load_b(Wt, n0 + 32 + lr, 1024, k0, half);
        v16bf b3 = load_b(Wt, n0 + 48 + lr, 1024, k0, half);
        c00 = wmma_bf16(a0, b0, c00);
        c10 = wmma_bf16(a1, b0, c10);
        c01 = wmma_bf16(a0, b1, c01);
        c11 = wmma_bf16(a1, b1, c11);
        c02 = wmma_bf16(a0, b2, c02);
        c12 = wmma_bf16(a1, b2, c12);
        c03 = wmma_bf16(a0, b3, c03);
        c13 = wmma_bf16(a1, b3, c13);
    }

    v8f cc[2][4] = {{c00, c01, c02, c03}, {c10, c11, c12, c13}};
    for (int i = 0; i < 2; ++i) {
        for (int j = 0; j < 4; ++j) {
            int n = n0 + j * 16 + lr;
            int h = n >> 6;
            int d = n & 63;
            float bvl = bias[n];
            for (int r = 0; r < 8; ++r) {
                int m = m0 + i * 16 + r + half * 8;
                int b = m >> 11;
                int s = m & 2047;
                unsigned short val = f2bfu(cc[i][j][r] + bvl);
                if (transposeV)
                    out[(((size_t)(b * 16 + h) * 64 + d) * 2048 + s)] = val;
                else
                    out[(((size_t)(b * 16 + h) * 2048 + s) * 64 + d)] = val;
            }
        }
    }
}

// ---------------------------------------------------------------------------
// Banded attention: one wave per (b, h, 16-query tile).
// 17 key tiles cover the [q-255, q] band; P padded to 288 cols (9 x K=32).
// ---------------------------------------------------------------------------
__global__ void attn_kernel(const unsigned short* __restrict__ Qb_,
                            const unsigned short* __restrict__ Kb_,
                            const unsigned short* __restrict__ Vt_,
                            unsigned short* __restrict__ Obf) {
    __shared__ __align__(16) float          sc[16 * 288];
    __shared__ __align__(16) unsigned short pb[16 * 288];

    const int S = 2048, Dh = 64;
    const int q0 = blockIdx.x * 16;
    const int h  = blockIdx.y;
    const int b  = blockIdx.z;
    const int lane = threadIdx.x;
    const int half = lane >> 4;
    const int lr = lane & 15;

    const unsigned short* Qb = Qb_ + (size_t)(b * 16 + h) * S * Dh;
    const unsigned short* Kb = Kb_ + (size_t)(b * 16 + h) * S * Dh;
    const unsigned short* Vt = Vt_ + (size_t)(b * 16 + h) * Dh * S;

    // Q fragments for this tile (reused across all key tiles)
    v16bf qa0 = load_a(Qb, q0 + lr, Dh, 0, half);
    v16bf qa1 = load_a(Qb, q0 + lr, Dh, 32, half);

    // ---- stage 1: masked scores into LDS ----
    for (int t = 0; t < 17; ++t) {
        int kt0  = q0 - 256 + t * 16;
        int krow = kt0 + lr;
        int krc  = min(max(krow, 0), S - 1);          // clamp OOB rows (masked anyway)
        v16bf kb0 = load_b(Kb, krc, Dh, 0, half);
        v16bf kb1 = load_b(Kb, krc, Dh, 32, half);
        v8f c = {0.f,0.f,0.f,0.f,0.f,0.f,0.f,0.f};
        c = wmma_bf16(qa0, kb0, c);
        c = wmma_bf16(qa1, kb1, c);
        for (int r = 0; r < 8; ++r) {
            int q = q0 + r + half * 8;
            int k = krow;
            int d = q - k;
            float sv = c[r] * 0.125f;                 // 1/sqrt(64)
            if (!(d >= 0 && d < 256 && k >= 0)) sv = -1e30f;
            sc[(r + half * 8) * 288 + t * 16 + lr] = sv;
        }
    }
    // pad tile 17 (cols 272..287) fully masked -> exact zeros after exp
    for (int r = 0; r < 8; ++r)
        sc[(r + half * 8) * 288 + 272 + lr] = -1e30f;
    __syncthreads();

    // ---- stage 2: softmax (lane = (row, half-segment of 144 cols)) ----
    {
        int row = lr;
        float* rowp = &sc[row * 288 + half * 144];
        float mx = -1e30f;
        for (int j = 0; j < 144; ++j) mx = fmaxf(mx, rowp[j]);
        mx = fmaxf(mx, __shfl_xor(mx, 16, 32));
        float sum = 0.f;
        for (int j = 0; j < 144; ++j) {
            float e = __expf(rowp[j] - mx);
            rowp[j] = e;
            sum += e;
        }
        sum += __shfl_xor(sum, 16, 32);
        float inv = 1.0f / sum;
        unsigned short* prow = &pb[row * 288 + half * 144];
        for (int j = 0; j < 144; ++j)
            prow[j] = f2bfu(rowp[j] * inv);
    }
    __syncthreads();

    // ---- stage 3: O = P @ V  (Vt rows are d, contiguous in s) ----
    int s_base = q0 - 256;
    for (int j = 0; j < 4; ++j) {
        v8f c = {0.f,0.f,0.f,0.f,0.f,0.f,0.f,0.f};
        for (int t = 0; t < 9; ++t) {
            v16bf pa = load_a(pb, lr, 288, t * 32, half);
            v16bf vb = load_b(Vt, j * 16 + lr, S, s_base + t * 32, half);
            c = wmma_bf16(pa, vb, c);
        }
        for (int r = 0; r < 8; ++r) {
            int s = q0 + r + half * 8;
            int n = h * 64 + j * 16 + lr;
            Obf[((size_t)b * 2048 + s) * 1024 + n] = f2bfu(c[r]);
        }
    }
}

// ---------------------------------------------------------------------------
// Output GEMM: Y = Obf[4096x1024] @ Wo + bo  (fp32 output), 32x64 wave tile
// ---------------------------------------------------------------------------
__global__ void out_gemm_kernel(const unsigned short* __restrict__ Obf,
                                const unsigned short* __restrict__ Wot,
                                const float* __restrict__ bo,
                                float* __restrict__ Y) {
    const int m0 = blockIdx.x * 32;
    const int n0 = blockIdx.y * 64;
    const int lane = threadIdx.x;
    const int half = lane >> 4;
    const int lr = lane & 15;

    v8f cz = {0.f,0.f,0.f,0.f,0.f,0.f,0.f,0.f};
    v8f c00 = cz, c01 = cz, c02 = cz, c03 = cz;
    v8f c10 = cz, c11 = cz, c12 = cz, c13 = cz;

    for (int k0 = 0; k0 < 1024; k0 += 32) {
        v16bf a0 = load_a(Obf, m0 +      lr, 1024, k0, half);
        v16bf a1 = load_a(Obf, m0 + 16 + lr, 1024, k0, half);
        v16bf b0 = load_b(Wot, n0 +  0 + lr, 1024, k0, half);
        v16bf b1 = load_b(Wot, n0 + 16 + lr, 1024, k0, half);
        v16bf b2 = load_b(Wot, n0 + 32 + lr, 1024, k0, half);
        v16bf b3 = load_b(Wot, n0 + 48 + lr, 1024, k0, half);
        c00 = wmma_bf16(a0, b0, c00);
        c10 = wmma_bf16(a1, b0, c10);
        c01 = wmma_bf16(a0, b1, c01);
        c11 = wmma_bf16(a1, b1, c11);
        c02 = wmma_bf16(a0, b2, c02);
        c12 = wmma_bf16(a1, b2, c12);
        c03 = wmma_bf16(a0, b3, c03);
        c13 = wmma_bf16(a1, b3, c13);
    }

    v8f cc[2][4] = {{c00, c01, c02, c03}, {c10, c11, c12, c13}};
    for (int i = 0; i < 2; ++i) {
        for (int j = 0; j < 4; ++j) {
            int n = n0 + j * 16 + lr;
            float bvl = bo[n];
            for (int r = 0; r < 8; ++r) {
                int m = m0 + i * 16 + r + half * 8;
                Y[(size_t)m * 1024 + n] = cc[i][j][r] + bvl;
            }
        }
    }
}

// ---------------------------------------------------------------------------
extern "C" void kernel_launch(void* const* d_in, const int* in_sizes, int n_in,
                              void* d_out, int out_size, void* d_ws, size_t ws_size,
                              hipStream_t stream) {
    (void)in_sizes; (void)n_in; (void)out_size; (void)ws_size;

    const float* X  = (const float*)d_in[0];
    const float* Wq = (const float*)d_in[1];
    const float* bq = (const float*)d_in[2];
    const float* Wk = (const float*)d_in[3];
    const float* bk = (const float*)d_in[4];
    const float* Wv = (const float*)d_in[5];
    const float* bv = (const float*)d_in[6];
    const float* Wo = (const float*)d_in[7];
    const float* bo = (const float*)d_in[8];
    float* Y = (float*)d_out;

    const size_t BS = 4096, D = 1024;

    // workspace layout (bf16 / ushort elements), total ~48 MB
    unsigned short* Xbf = (unsigned short*)d_ws;
    unsigned short* Wqt = Xbf + BS * D;
    unsigned short* Wkt = Wqt + D * D;
    unsigned short* Wvt = Wkt + D * D;
    unsigned short* Wot = Wvt + D * D;
    unsigned short* Qbf = Wot + D * D;   // [B,H,S,Dh]
    unsigned short* Kbf = Qbf + BS * D;  // [B,H,S,Dh]
    unsigned short* Vt  = Kbf + BS * D;  // [B,H,Dh,S]
    unsigned short* Obf = Vt  + BS * D;  // [B,S,D]

    cvt_x_kernel <<<(int)((BS * D) / 256), 256, 0, stream>>>(X, Xbf, (int)(BS * D));
    cvt_wT_kernel<<<(int)((D * D) / 256), 256, 0, stream>>>(Wq, Wqt);
    cvt_wT_kernel<<<(int)((D * D) / 256), 256, 0, stream>>>(Wk, Wkt);
    cvt_wT_kernel<<<(int)((D * D) / 256), 256, 0, stream>>>(Wv, Wvt);
    cvt_wT_kernel<<<(int)((D * D) / 256), 256, 0, stream>>>(Wo, Wot);

    dim3 gg(128, 16);   // 4096/32 M-tiles x 1024/64 N-tiles, 1 wave each
    proj_gemm_kernel<<<gg, 32, 0, stream>>>(Xbf, Wqt, bq, Qbf, 0);
    proj_gemm_kernel<<<gg, 32, 0, stream>>>(Xbf, Wkt, bk, Kbf, 0);
    proj_gemm_kernel<<<gg, 32, 0, stream>>>(Xbf, Wvt, bv, Vt, 1);

    dim3 ga(128, 16, 2); // S/16 q-tiles x H x B
    attn_kernel<<<ga, 32, 0, stream>>>(Qbf, Kbf, Vt, Obf);

    out_gemm_kernel<<<gg, 32, 0, stream>>>(Obf, Wot, bo, Y);
}
